// Attention_10359461118430
// MI455X (gfx1250) — compile-verified, matrix operand order checked by reading, not run on hardware
//
#include <hip/hip_runtime.h>
#include <hip/hip_bf16.h>

#define BS  4
#define DL  1024
#define NX  1024
#define NH  16
#define DH  64
#define KNN 64

typedef __attribute__((ext_vector_type(16))) __bf16 v16bf;
typedef __attribute__((ext_vector_type(8)))  __bf16 bf16x8;
typedef __attribute__((ext_vector_type(8)))  float  v8f;

// ---------------------------------------------------------------------------
// Fragment loader: A(16x32) / B(32x16 from n-major operand) bf16 fragment.
// Per ISA layout: lane m|n = lane&15, half = lane>>4,
// element e -> k = (e<8 ? 0 : 16) + 8*half + (e&7).
// => two contiguous 8x bf16 (16B) loads.
// ---------------------------------------------------------------------------
__device__ __forceinline__ v16bf load_frag(const __bf16* p) {
    bf16x8 lo = *(const bf16x8*)(p);
    bf16x8 hi = *(const bf16x8*)(p + 16);
    return __builtin_shufflevector(lo, hi, 0,1,2,3,4,5,6,7,8,9,10,11,12,13,14,15);
}

__device__ __forceinline__ v8f wmma_bf16(v16bf a, v16bf b, v8f c) {
    return __builtin_amdgcn_wmma_f32_16x16x32_bf16(false, a, false, b, (short)0, c, false, false);
}

// ---------------------------------------------------------------------------
// Prep kernels: fp32 -> bf16 copies / transposes, mask bitset
// ---------------------------------------------------------------------------
__global__ void k_convert_x(const float* __restrict__ x, __bf16* __restrict__ xb, int n) {
    int i = blockIdx.x * blockDim.x + threadIdx.x;
    if (i < n) xb[i] = (__bf16)x[i];
}

// WT[n][k] = W[k][n], W is (NX x 3NX)
__global__ void k_transpose_wattn(const float* __restrict__ W, __bf16* __restrict__ WT) {
    int i = blockIdx.x * blockDim.x + threadIdx.x;
    if (i >= NX * 3 * NX) return;
    int n = i / NX, k = i % NX;
    WT[n * NX + k] = (__bf16)W[k * (3 * NX) + n];
}

// WpT[n][k] = Wp[k][n], Wp is (NX x NX)
__global__ void k_transpose_wproj(const float* __restrict__ W, __bf16* __restrict__ WT) {
    int i = blockIdx.x * blockDim.x + threadIdx.x;
    if (i >= NX * NX) return;
    int n = i / NX, k = i % NX;
    WT[n * NX + k] = (__bf16)W[k * NX + n];
}

__global__ void k_zero_bits(unsigned int* __restrict__ bits, int nwords) {
    int i = blockIdx.x * blockDim.x + threadIdx.x;
    if (i < nwords) bits[i] = 0u;
}

// bits[b][i][j] : 1 if j in rns_indices[b,i,:]
__global__ void k_build_bits(const int* __restrict__ rns, unsigned int* __restrict__ bits) {
    int idx = blockIdx.x * blockDim.x + threadIdx.x;
    if (idx >= BS * DL * KNN) return;
    int b = idx / (DL * KNN);
    int i = (idx / KNN) % DL;
    int j = rns[idx] & (DL - 1);
    atomicOr(&bits[(b * DL + i) * (DL / 32) + (j >> 5)], 1u << (j & 31));
}

// ---------------------------------------------------------------------------
// QKV GEMM: (BS*DL x NX) @ (NX x 3NX) + bias.
// Each wave: one 16-row strip x 64 cols (4 accumulators), K-loop step 32.
// Epilogue scatters: q,k -> [b][h][i][d] bf16 ; v -> transposed [b][h][d][i].
// ---------------------------------------------------------------------------
__global__ void k_qkv(const __bf16* __restrict__ xb, const __bf16* __restrict__ WT,
                      const float* __restrict__ b_attn,
                      __bf16* __restrict__ qb, __bf16* __restrict__ kb,
                      __bf16* __restrict__ vT) {
    const int lane  = threadIdx.x & 31;
    const int lin   = (blockIdx.x * blockDim.x + threadIdx.x) >> 5;
    const int NGRP  = (3 * NX) / 64;                    // 48
    if (lin >= (BS * DL / 16) * NGRP) return;
    const int mtile = lin / NGRP;
    const int ng    = lin % NGRP;
    const int l16   = lane & 15;
    const int h8    = (lane >> 4) * 8;

    const __bf16* arow = xb + (size_t)(mtile * 16 + l16) * NX;
    const __bf16* brow[4];
#pragma unroll
    for (int t = 0; t < 4; ++t)
        brow[t] = WT + (size_t)(ng * 64 + t * 16 + l16) * NX;

    v8f acc[4] = {v8f{}, v8f{}, v8f{}, v8f{}};
    for (int kk = 0; kk < NX; kk += 32) {
        __builtin_prefetch(arow + kk + 256, 0, 1);      // global_prefetch_b8
        v16bf aF = load_frag(arow + kk + h8);
#pragma unroll
        for (int t = 0; t < 4; ++t)
            acc[t] = wmma_bf16(aF, load_frag(brow[t] + kk + h8), acc[t]);
    }

    const int sec = ng / 16;                            // 0=q 1=k 2=v
#pragma unroll
    for (int t = 0; t < 4; ++t) {
        const int within = (ng % 16) * 64 + t * 16 + l16;   // 0..1023
        const int h = within >> 6, d = within & 63;
        const float bias = b_attn[sec * NX + within];
#pragma unroll
        for (int v = 0; v < 8; ++v) {
            const int row = mtile * 16 + v + h8;            // global M row
            const int b = row >> 10, i = row & (DL - 1);
            const float val = acc[t][v] + bias;
            const size_t hd = ((size_t)(b * NH + h));
            if (sec == 0)      qb[(hd * DL + i) * DH + d] = (__bf16)val;
            else if (sec == 1) kb[(hd * DL + i) * DH + d] = (__bf16)val;
            else               vT[(hd * DH + d) * DL + i] = (__bf16)val;
        }
    }
}

// ---------------------------------------------------------------------------
// Fused masked-softmax attention (flash style).
// One wave owns a 16-row i-tile of one (b,h): loops j in steps of 32.
// Per iter: 4 score WMMAs (2 j-subtiles x K=64), bit-mask + -1e9 fill,
// online softmax, p-tile -> LDS (bf16) -> A-fragment, 4 p.v WMMAs.
// ---------------------------------------------------------------------------
__global__ void k_flash(const __bf16* __restrict__ qb, const __bf16* __restrict__ kb,
                        const __bf16* __restrict__ vT, const unsigned int* __restrict__ bits,
                        __bf16* __restrict__ ab) {
    __shared__ __bf16 ldsP[4 * 16 * 32];                // 4 waves/block, 16x32 each

    const int lane = threadIdx.x & 31;
    const int wid  = threadIdx.x >> 5;
    const int lin  = blockIdx.x * 4 + wid;
    if (lin >= BS * NH * (DL / 16)) return;
    const int itile = lin % (DL / 16);
    const int bh    = lin / (DL / 16);
    const int b = bh / NH, h = bh % NH;
    const int iBase = itile * 16;
    const int l16 = lane & 15, half = lane >> 4, h8 = half * 8;

    __bf16* myP = ldsP + wid * 16 * 32;
    const size_t bhOff = (size_t)(b * NH + h);

    // q A-fragments (K = DH = 64 -> two k-steps), invariant over j loop
    const __bf16* qrow = qb + (bhOff * DL + iBase + l16) * DH;
    v16bf qA0 = load_frag(qrow + 0  + h8);
    v16bf qA1 = load_frag(qrow + 32 + h8);

    const __bf16* vrow[4];
#pragma unroll
    for (int t = 0; t < 4; ++t)
        vrow[t] = vT + (bhOff * DH + t * 16 + l16) * DL;

    v8f   accO[4] = {v8f{}, v8f{}, v8f{}, v8f{}};
    float rmax[8], rsum[8];
#pragma unroll
    for (int v = 0; v < 8; ++v) { rmax[v] = -INFINITY; rsum[v] = 0.0f; }

    const int colWord = iBase >> 5;

    for (int j0 = 0; j0 < DL; j0 += 32) {
        // ---- scores: two 16x16 tiles over j, K=64
        v8f s[2] = {v8f{}, v8f{}};
#pragma unroll
        for (int t = 0; t < 2; ++t) {
            const __bf16* krow = kb + (bhOff * DL + j0 + t * 16 + l16) * DH;
            s[t] = wmma_bf16(qA0, load_frag(krow + 0  + h8), s[t]);
            s[t] = wmma_bf16(qA1, load_frag(krow + 32 + h8), s[t]);
        }

        // ---- symmetric sparse mask + NEG fill
        unsigned wrow[8];
#pragma unroll
        for (int v = 0; v < 8; ++v)
            wrow[v] = bits[(size_t)(b * DL + iBase + v + h8) * 32 + (j0 >> 5)];
        float val[2][8], tmax[8];
#pragma unroll
        for (int v = 0; v < 8; ++v) tmax[v] = -INFINITY;
#pragma unroll
        for (int t = 0; t < 2; ++t) {
            const int j = j0 + t * 16 + l16;
            const unsigned wcol = bits[(size_t)(b * DL + j) * 32 + colWord];
#pragma unroll
            for (int v = 0; v < 8; ++v) {
                const int ir = iBase + v + h8;
                const unsigned mbit = ((wrow[v] >> (j & 31)) & 1u) & ((wcol >> (ir & 31)) & 1u);
                const float mk = (float)mbit;
                const float sv = s[t][v] * 0.125f;              // 1/sqrt(DH)
                val[t][v] = sv * mk + (-1.0e9f) * (1.0f - mk);
                tmax[v] = fmaxf(tmax[v], val[t][v]);
            }
        }
        // ---- row max across the 16 lanes of this half (n-dimension)
#pragma unroll
        for (int off = 1; off < 16; off <<= 1)
#pragma unroll
            for (int v = 0; v < 8; ++v)
                tmax[v] = fmaxf(tmax[v], __shfl_xor(tmax[v], off, 32));

        float corr[8], psum[8];
#pragma unroll
        for (int v = 0; v < 8; ++v) {
            const float nm = fmaxf(rmax[v], tmax[v]);
            corr[v] = __expf(rmax[v] - nm);
            rmax[v] = nm;
            psum[v] = 0.0f;
        }
        // ---- p = exp(val - max), write p-tile to LDS in row-major bf16
        float pv[2][8];
#pragma unroll
        for (int t = 0; t < 2; ++t)
#pragma unroll
            for (int v = 0; v < 8; ++v) {
                const float p = __expf(val[t][v] - rmax[v]);
                pv[t][v] = p;
                psum[v] += p;
                myP[(v + h8) * 32 + t * 16 + l16] = (__bf16)p;
            }
#pragma unroll
        for (int off = 1; off < 16; off <<= 1)
#pragma unroll
            for (int v = 0; v < 8; ++v)
                psum[v] += __shfl_xor(psum[v], off, 32);
#pragma unroll
        for (int v = 0; v < 8; ++v) {
            rsum[v] = rsum[v] * corr[v] + psum[v];
#pragma unroll
            for (int t = 0; t < 4; ++t) accO[t][v] *= corr[v];
        }

        // cross-lane LDS store->load inside one wave: drain DS, fence scheduling
        asm volatile("s_wait_dscnt 0" ::: "memory");
        __builtin_amdgcn_wave_barrier();

        // ---- p A-fragment (16 rows x 32 j) from LDS, then p.v WMMAs
        v16bf pA = load_frag(myP + l16 * 32 + h8);
#pragma unroll
        for (int t = 0; t < 4; ++t)
            accO[t] = wmma_bf16(pA, load_frag(vrow[t] + j0 + h8), accO[t]);
    }

    // ---- normalize and store a[b][i][h*DH + d] as bf16
#pragma unroll
    for (int t = 0; t < 4; ++t) {
        const int d = t * 16 + l16;
#pragma unroll
        for (int v = 0; v < 8; ++v) {
            const int i = iBase + v + h8;
            ab[((size_t)(b * DL + i)) * NX + h * DH + d] = (__bf16)(accO[t][v] / rsum[v]);
        }
    }
}

// ---------------------------------------------------------------------------
// Output projection: (BS*DL x NX) @ (NX x NX) + bias -> fp32 out
// ---------------------------------------------------------------------------
__global__ void k_proj(const __bf16* __restrict__ ab, const __bf16* __restrict__ WpT,
                       const float* __restrict__ b_proj, float* __restrict__ out) {
    const int lane  = threadIdx.x & 31;
    const int lin   = (blockIdx.x * blockDim.x + threadIdx.x) >> 5;
    const int NGRP  = NX / 64;                          // 16
    if (lin >= (BS * DL / 16) * NGRP) return;
    const int mtile = lin / NGRP;
    const int ng    = lin % NGRP;
    const int l16   = lane & 15;
    const int h8    = (lane >> 4) * 8;

    const __bf16* arow = ab + (size_t)(mtile * 16 + l16) * NX;
    const __bf16* brow[4];
#pragma unroll
    for (int t = 0; t < 4; ++t)
        brow[t] = WpT + (size_t)(ng * 64 + t * 16 + l16) * NX;

    v8f acc[4] = {v8f{}, v8f{}, v8f{}, v8f{}};
    for (int kk = 0; kk < NX; kk += 32) {
        __builtin_prefetch(arow + kk + 256, 0, 1);
        v16bf aF = load_frag(arow + kk + h8);
#pragma unroll
        for (int t = 0; t < 4; ++t)
            acc[t] = wmma_bf16(aF, load_frag(brow[t] + kk + h8), acc[t]);
    }
#pragma unroll
    for (int t = 0; t < 4; ++t) {
        const int n = ng * 64 + t * 16 + l16;
        const float bias = b_proj[n];
#pragma unroll
        for (int v = 0; v < 8; ++v) {
            const int row = mtile * 16 + v + h8;
            out[(size_t)row * NX + n] = acc[t][v] + bias;
        }
    }
}

// ---------------------------------------------------------------------------
// Host launcher
// ---------------------------------------------------------------------------
extern "C" void kernel_launch(void* const* d_in, const int* in_sizes, int n_in,
                              void* d_out, int out_size, void* d_ws, size_t ws_size,
                              hipStream_t stream) {
    const float* x      = (const float*)d_in[0];
    const int*   rns    = (const int*)  d_in[2];
    const float* W_attn = (const float*)d_in[3];
    const float* b_attn = (const float*)d_in[4];
    const float* W_proj = (const float*)d_in[5];
    const float* b_proj = (const float*)d_in[6];
    float* out = (float*)d_out;

    // workspace carve-up (bytes)
    char* ws = (char*)d_ws;
    __bf16* xb  = (__bf16*)(ws);                                   //  8 MB
    __bf16* WT  = (__bf16*)(ws + (size_t)8  * 1024 * 1024);        //  6 MB
    __bf16* WpT = (__bf16*)(ws + (size_t)14 * 1024 * 1024);        //  2 MB
    __bf16* qb  = (__bf16*)(ws + (size_t)16 * 1024 * 1024);        //  8 MB
    __bf16* kb  = (__bf16*)(ws + (size_t)24 * 1024 * 1024);        //  8 MB
    __bf16* vT  = (__bf16*)(ws + (size_t)32 * 1024 * 1024);        //  8 MB
    __bf16* ab  = (__bf16*)(ws + (size_t)40 * 1024 * 1024);        //  8 MB
    unsigned int* bits = (unsigned int*)(ws + (size_t)48 * 1024 * 1024); // 512 KB

    const int nx_elems = BS * DL * NX;
    k_convert_x     <<<nx_elems / 256, 256, 0, stream>>>(x, xb, nx_elems);
    k_transpose_wattn<<<(NX * 3 * NX) / 256, 256, 0, stream>>>(W_attn, WT);
    k_transpose_wproj<<<(NX * NX) / 256, 256, 0, stream>>>(W_proj, WpT);

    const int nwords = BS * DL * (DL / 32);
    k_zero_bits <<<nwords / 256, 256, 0, stream>>>(bits, nwords);
    k_build_bits<<<(BS * DL * KNN) / 256, 256, 0, stream>>>(rns, bits);

    // QKV: 256 m-tiles x 48 n-groups = 12288 waves, 8 waves/block
    k_qkv<<<1536, 256, 0, stream>>>(xb, WT, b_attn, qb, kb, vT);

    // flash attention: 4096 i-tiles, 4 waves/block
    k_flash<<<1024, 128, 0, stream>>>(qb, kb, vT, bits, ab);

    // projection: 256 x 16 = 4096 waves, 8 waves/block
    k_proj<<<512, 256, 0, stream>>>(ab, WpT, b_proj, out);
}
